// CostLearnerNet_35424890258182
// MI455X (gfx1250) — matrix operand order repaired and checked by reading, not compiled, not run on hardware
//
#include <hip/hip_runtime.h>
#include <math.h>

typedef __attribute__((ext_vector_type(16))) _Float16 v16h;
typedef __attribute__((ext_vector_type(8)))  _Float16 v8h;
typedef __attribute__((ext_vector_type(2)))  _Float16 v2h;
typedef __attribute__((ext_vector_type(8)))  float    v8f;

#define T_ 512
#define F_ 256

// ---- workspace layout (bytes) ----
// xg^T fragments (f16, WMMA C/D order, transposed form): [t][bg][gtile][lane][8 halfs]
#define XG_OFF    0ull
#define XG_BYTES  (512ull * 32 * 8 * 32 * 16)          // 64 MB
// Wih^T A-fragments (f16): [kstep(8)][gtile(8)][lane(32)][16 halfs]
#define WIH_OFF   (XG_OFF + XG_BYTES)
#define WIH_BYTES (8ull * 8 * 32 * 16 * 2)             // 64 KB
// Whh^T A-fragments (f16): [gtile(8)][lane(32)][16 halfs]
#define WHH_OFF   (WIH_OFF + WIH_BYTES)
#define WHH_BYTES (8ull * 32 * 16 * 2)                 // 8 KB
// sampled bias (f32, 128 -> pad 512B)
#define BS_OFF    (WHH_OFF + WHH_BYTES)
// final hidden state h_last (f32) [512][32]
#define HL_OFF    (BS_OFF + 512ull)

__device__ __forceinline__ float softplusf(float x) {
  return log1pf(__expf(x));
}

__device__ __forceinline__ float fast_tanh(float x) {
#if __has_builtin(__builtin_amdgcn_tanhf)
  return __builtin_amdgcn_tanhf(x);
#elif __has_builtin(__builtin_amdgcn_tanh_f32)
  return __builtin_amdgcn_tanh_f32(x);
#else
  float r;
  asm("v_tanh_f32 %0, %1" : "=v"(r) : "v"(x));
  return r;
#endif
}

__device__ __forceinline__ float fast_sigmoid(float x) {
  return __builtin_fmaf(0.5f, fast_tanh(0.5f * x), 0.5f);
}

__device__ __forceinline__ unsigned pack_h2(float a, float b) {
  v2h t;
  t[0] = (_Float16)a;
  t[1] = (_Float16)b;
  return __builtin_bit_cast(unsigned, t);   // v_cvt_pk_f16_f32
}

// A-fragment scatter position for a 16x32 f16 A tile (contiguous k order):
// k 0..31 ->  hi = k>>4, slot = k&15;  lane = row%16 + 16*hi
// B-fragment: lanes 0-15 hold K=0..15 (slot=K), lanes 16-31 hold K=16..31.

// ------------------------------------------------------------------
// Phase 0: sample Bayesian weights, scatter into WMMA A-fragment order
// (transposed form: A = W^T, rows = gate index g, cols = K)
// ------------------------------------------------------------------
__global__ void k_sample(const float* __restrict__ wih_mu, const float* __restrict__ wih_rho,
                         const float* __restrict__ eps_ih,
                         const float* __restrict__ whh_mu, const float* __restrict__ whh_rho,
                         const float* __restrict__ eps_hh,
                         const float* __restrict__ b_mu, const float* __restrict__ b_rho,
                         const float* __restrict__ eps_b,
                         _Float16* __restrict__ wih_frag, _Float16* __restrict__ whh_frag,
                         float* __restrict__ b_s) {
  int idx = blockIdx.x * blockDim.x + threadIdx.x;
  if (idx < 32768) {                       // Wih [256 f][128 g]
    float w = wih_mu[idx] + softplusf(wih_rho[idx]) * eps_ih[idx];
    int f = idx >> 7, g = idx & 127;
    int ks = f >> 5, kin = f & 31;
    int gt = g >> 4, mrow = g & 15;
    int lane = mrow + 16 * (kin >> 4);
    int slot = kin & 15;
    wih_frag[(((ks * 8 + gt) * 32 + lane) << 4) + slot] = (_Float16)w;
  } else if (idx < 32768 + 4096) {         // Whh [32 k][128 g]
    int i2 = idx - 32768;
    float w = whh_mu[i2] + softplusf(whh_rho[i2]) * eps_hh[i2];
    int k = i2 >> 7, g = i2 & 127;
    int gt = g >> 4, mrow = g & 15;
    int lane = mrow + 16 * (k >> 4);
    int slot = k & 15;
    whh_frag[((gt * 32 + lane) << 4) + slot] = (_Float16)w;
  } else if (idx < 32768 + 4096 + 128) {   // bias [128]
    int g = idx - 32768 - 4096;
    b_s[g] = b_mu[g] + softplusf(b_rho[g]) * eps_b[g];
  }
}

// ------------------------------------------------------------------
// Phase 1: xg^T[t,bg] = Wih^T (A) x x^T (B) + bias, stored as f16 D-fragments.
// One wave per (t, batch-group of 16); 64 WMMAs per wave.
// B-frag of x^T: lane holds 16 *contiguous* floats of one x row -> 4x b128.
// ------------------------------------------------------------------
__global__ __launch_bounds__(256) void k_xproj(const float* __restrict__ x,
                                               const _Float16* __restrict__ wih_frag,
                                               const float* __restrict__ b_s,
                                               v8h* __restrict__ xg_frag) {
  int wid = (blockIdx.x * 256 + (int)threadIdx.x) >> 5;   // 0..16383
  int lid = threadIdx.x & 31;
  int t  = wid >> 5;                                      // 0..511
  int bg = wid & 31;                                      // 0..31
  int bat = lid & 15, hi = lid >> 4;

  const float* xrow = x + ((size_t)(bg * 16 + bat) * T_ + t) * F_;
  const v16h* af = (const v16h*)wih_frag;

  v8f acc[8];
  #pragma unroll
  for (int gt = 0; gt < 8; ++gt)
    #pragma unroll
    for (int e = 0; e < 8; ++e) acc[gt][e] = 0.0f;

  #pragma unroll 1
  for (int ks = 0; ks < 8; ++ks) {
    // B fragment: col = batch (lane%16), K slot s <-> f = ks*32 + 16*hi + s
    v16h b;
    const float4* xp = (const float4*)(xrow + ks * 32 + 16 * hi);
    #pragma unroll
    for (int p = 0; p < 4; ++p) {
      float4 f4 = xp[p];
      b[4 * p + 0] = (_Float16)f4.x;
      b[4 * p + 1] = (_Float16)f4.y;
      b[4 * p + 2] = (_Float16)f4.z;
      b[4 * p + 3] = (_Float16)f4.w;
    }
    #pragma unroll
    for (int gt = 0; gt < 8; ++gt) {
      v16h a = af[(ks * 8 + gt) * 32 + lid];
      acc[gt] = __builtin_amdgcn_wmma_f32_16x16x32_f16(false, a, false, b,
                                                       (short)0, acc[gt], false, false);
    }
  }

  // + bias (g = gt*16 + 8*hi + e), convert, store fragments (one b128 each)
  size_t base = (size_t)(t * 32 + bg) * 8 * 32;
  #pragma unroll
  for (int gt = 0; gt < 8; ++gt) {
    v8h o;
    #pragma unroll
    for (int e = 0; e < 8; ++e)
      o[e] = (_Float16)(acc[gt][e] + b_s[gt * 16 + 8 * hi + e]);
    xg_frag[base + gt * 32 + lid] = o;
  }
}

// ------------------------------------------------------------------
// Phase 2: LSTM recurrence, transposed form. 2 waves per 16-row batch group.
// Wave q owns gate tiles {q, q+2, q+4, q+6} and hidden block j in [16q,16q+16).
// Per step: 4 WMMA + 8 elems/lane of gate math (v_tanh) + 1 LDS b128 store +
// one 32B LDS load + 1 barrier (double-buffered h).
// ------------------------------------------------------------------
__global__ __launch_bounds__(64) void k_scan(const v8h* __restrict__ xg_frag,
                                             const _Float16* __restrict__ whh_frag,
                                             float* __restrict__ h_last) {
  __shared__ __align__(32) unsigned hbuf[2][256];   // [buf][512 halfs]: h[batch 16][k 32]
  int tid = threadIdx.x;
  int lid = tid & 31;
  int q   = tid >> 5;                 // wave id = hidden block
  int bg  = blockIdx.x;
  int bat = lid & 15, hi = lid >> 4;

  // Whh^T A-fragments for this wave's 4 gate tiles
  const v16h* af = (const v16h*)whh_frag;
  v16h wa[4];
  #pragma unroll
  for (int i = 0; i < 4; ++i) wa[i] = af[(2 * i + q) * 32 + lid];

  float cs[8];
  #pragma unroll
  for (int v = 0; v < 8; ++v) cs[v] = 0.0f;

  // zero initial read buffer (h0 = 0)
  #pragma unroll
  for (int i = 0; i < 4; ++i) hbuf[0][tid * 4 + i] = 0u;
  __syncthreads();

  const unsigned wr_u = (unsigned)(bat * 16 + q * 8 + 4 * hi);  // uint idx of h write
  const unsigned rd_u = (unsigned)(bat * 16 + 8 * hi);          // uint idx of B-frag read

  #pragma unroll 1
  for (int t = 0; t < 512; ++t) {
    int rb = t & 1, wb = rb ^ 1;
    // C init from xg^T fragments (coalesced b128 loads)
    v8f acc[4];
    size_t base = (size_t)(t * 32 + bg) * 8 * 32;
    #pragma unroll
    for (int i = 0; i < 4; ++i) {
      v8h xh = xg_frag[base + (2 * i + q) * 32 + lid];
      #pragma unroll
      for (int e = 0; e < 8; ++e) acc[i][e] = (float)xh[e];
    }
    // prefetch next timestep's fragments (global_prefetch_b8)
    if (t + 1 < 512) {
      size_t nbase = (size_t)((t + 1) * 32 + bg) * 8 * 32;
      #pragma unroll
      for (int i = 0; i < 4; ++i)
        __builtin_prefetch((const void*)&xg_frag[nbase + (2 * i + q) * 32 + lid], 0, 0);
    }
    // B fragment of h^T: lane holds h[batch][16*hi .. 16*hi+15] contiguously
    v16h hb = *(const v16h*)&hbuf[rb][rd_u];
    #pragma unroll
    for (int i = 0; i < 4; ++i)
      acc[i] = __builtin_amdgcn_wmma_f32_16x16x32_f16(false, wa[i], false, hb,
                                                      (short)0, acc[i], false, false);
    // gate math: lane owns (batch, j = q*16 + 8*hi + v), v = 0..7
    float hv[8];
    #pragma unroll
    for (int v = 0; v < 8; ++v) {
      float ig = fast_sigmoid(acc[0][v]);   // tile q      -> input gate
      float fg = fast_sigmoid(acc[1][v]);   // tile q+2    -> forget gate
      float gg = fast_tanh   (acc[2][v]);   // tile q+4    -> cell gate
      float og = fast_sigmoid(acc[3][v]);   // tile q+6    -> output gate
      float c  = __builtin_fmaf(fg, cs[v], ig * gg);
      cs[v] = c;
      hv[v] = og * fast_tanh(c);
    }
    if (t == 511) {
      #pragma unroll
      for (int v = 0; v < 8; ++v)
        h_last[(bg * 16 + bat) * 32 + q * 16 + 8 * hi + v] = hv[v];
    }
    // pack 8 h values -> 4 half2 words, one ds_store_b128
    uint4 pk = make_uint4(pack_h2(hv[0], hv[1]), pack_h2(hv[2], hv[3]),
                          pack_h2(hv[4], hv[5]), pack_h2(hv[6], hv[7]));
    *(uint4*)&hbuf[wb][wr_u] = pk;
    __syncthreads();
  }
}

// ------------------------------------------------------------------
// Phase 3: tiny MLP head: relu(h) -> relu(@w1+b1) -> @w2+b2
// ------------------------------------------------------------------
__global__ void k_head(const float* __restrict__ h_last,
                       const float* __restrict__ w1, const float* __restrict__ b1,
                       const float* __restrict__ w2, const float* __restrict__ b2,
                       float* __restrict__ out) {
  int b = blockIdx.x * blockDim.x + threadIdx.x;
  if (b >= 512) return;
  float hr[32];
  #pragma unroll
  for (int k = 0; k < 32; ++k) {
    float v = h_last[b * 32 + k];
    hr[k] = v > 0.0f ? v : 0.0f;
  }
  float acc = b2[0];
  #pragma unroll 1
  for (int j = 0; j < 16; ++j) {
    float s = b1[j];
    #pragma unroll
    for (int k = 0; k < 32; ++k) s += hr[k] * w1[k * 16 + j];
    s = s > 0.0f ? s : 0.0f;
    acc += s * w2[j];
  }
  out[b] = acc;
}

extern "C" void kernel_launch(void* const* d_in, const int* in_sizes, int n_in,
                              void* d_out, int out_size, void* d_ws, size_t ws_size,
                              hipStream_t stream) {
  (void)in_sizes; (void)n_in; (void)out_size; (void)ws_size;
  const float* x       = (const float*)d_in[0];
  const float* wih_mu  = (const float*)d_in[1];
  const float* wih_rho = (const float*)d_in[2];
  const float* eps_ih  = (const float*)d_in[3];
  const float* whh_mu  = (const float*)d_in[4];
  const float* whh_rho = (const float*)d_in[5];
  const float* eps_hh  = (const float*)d_in[6];
  const float* b_mu    = (const float*)d_in[7];
  const float* b_rho   = (const float*)d_in[8];
  const float* eps_b   = (const float*)d_in[9];
  const float* w1      = (const float*)d_in[10];
  const float* b1      = (const float*)d_in[11];
  const float* w2      = (const float*)d_in[12];
  const float* b2      = (const float*)d_in[13];

  char* ws = (char*)d_ws;
  v8h*      xg_frag  = (v8h*)(ws + XG_OFF);
  _Float16* wih_frag = (_Float16*)(ws + WIH_OFF);
  _Float16* whh_frag = (_Float16*)(ws + WHH_OFF);
  float*    b_s      = (float*)(ws + BS_OFF);
  float*    h_last   = (float*)(ws + HL_OFF);

  k_sample<<<(36992 + 255) / 256, 256, 0, stream>>>(wih_mu, wih_rho, eps_ih,
                                                    whh_mu, whh_rho, eps_hh,
                                                    b_mu, b_rho, eps_b,
                                                    wih_frag, whh_frag, b_s);
  k_xproj<<<2048, 256, 0, stream>>>(x, wih_frag, b_s, xg_frag);   // 16384 waves
  k_scan<<<32, 64, 0, stream>>>(xg_frag, whh_frag, h_last);       // 64 waves, 2/group
  k_head<<<2, 256, 0, stream>>>(h_last, w1, b1, w2, b2, (float*)d_out);
}
